// IdScaleAttn_56676388438017
// MI455X (gfx1250) — compile-verified, matrix operand order checked by reading, not run on hardware
//
#include <hip/hip_runtime.h>
#include <hip/hip_bf16.h>

typedef __attribute__((ext_vector_type(16))) _Float16 v16h;
typedef __attribute__((ext_vector_type(8)))  float    v8f;

#define FDIM 256
#define NH   8
#define NS   16   // M*4 corner samples

// ---- WMMA fragment loader from LDS (ISA 7.12.2 16-bit A layout) ----
// lanes 0-15 : elements 0..7 = K 0..7,  elements 8..15 = K 16..23
// lanes 16-31: elements 0..7 = K 8..15, elements 8..15 = K 24..31
__device__ __forceinline__ v16h frag_from_h(const _Float16* line, int lh) {
    v16h r; const int kb = lh * 8;
#pragma unroll
    for (int j = 0; j < 8; ++j) { r[j] = line[kb + j]; r[j + 8] = line[16 + kb + j]; }
    return r;
}

// ---------------- Kernel 0: pre-swizzle f32 weights into f16 B-fragments ----------
// dst layout: frag[tile(16)][kk(8)][lane(32)][e(16)], tile = 16 consecutive weight
// rows (= 16 output columns), kk = K-chunk of 32. Lane L holds weight row
// tile*16 + (L&15); element e maps to k = kk*32 + (e&8 ? 16:0) + (L>>4)*8 + (e&7).
__global__ void __launch_bounds__(256)
prep_wfrag_kernel(const float* __restrict__ w, _Float16* __restrict__ dst) {
    const int o  = blockIdx.x * 256 + threadIdx.x;      // 65536 elements
    const int e  = o & 15;
    const int L  = (o >> 4) & 31;
    const int kk = (o >> 9) & 7;
    const int ct = o >> 12;
    const int row = ct * 16 + (L & 15);
    const int k   = kk * 32 + ((e & 8) ? 16 : 0) + (L >> 4) * 8 + (e & 7);
    dst[o] = (_Float16)w[row * FDIM + k];
}

// ---------------- Kernel 1: attention + sampling metadata ----------------
// one wave32 per n: 32 logits (one per lane), softmax over groups of 4 (maps),
// lanes 0..15 compute bilinear weights + id-map gather for the 16 (m,s) samples.
__global__ void __launch_bounds__(256)
attn_meta_kernel(const float* __restrict__ act,  const int* __restrict__ bids,
                 const int* __restrict__ mids,   const int* __restrict__ xys,
                 const float* __restrict__ sembed, const float* __restrict__ attw,
                 const float* __restrict__ attb,
                 const int* __restrict__ im0, const int* __restrict__ im1,
                 const int* __restrict__ im2, const int* __restrict__ im3,
                 float* __restrict__ fw, int* __restrict__ fids, int n_total) {
    __shared__ float sact[8][FDIM];
    const int wave = threadIdx.x >> 5, lane = threadIdx.x & 31;
    const int n = blockIdx.x * 8 + wave;
    const bool valid = (n < n_total);

    int mid = 0, bid = 0; float px = 0.f, py = 0.f;
    if (valid) {
        mid = mids[n]; bid = bids[n];
        px = (float)xys[2 * n + 0];      // X
        py = (float)xys[2 * n + 1];      // Y
#pragma unroll
        for (int j = 0; j < 8; ++j) {
            int f = lane * 8 + j;
            sact[wave][f] = act[(size_t)n * FDIM + f] + sembed[mid * FDIM + f];
        }
    }
    __syncthreads();

    float logit = 0.f;
    if (valid) {
        const float* wrow = attw + lane * FDIM;
        float acc = 0.f;
        for (int f = 0; f < FDIM; f += 4) {
            acc += sact[wave][f + 0] * wrow[f + 0] + sact[wave][f + 1] * wrow[f + 1]
                 + sact[wave][f + 2] * wrow[f + 2] + sact[wave][f + 3] * wrow[f + 3];
        }
        logit = acc + attb[lane];
    }
    // softmax over 4-lane groups (logit index = h*4 + m)
    float mx = fmaxf(logit, __shfl_xor(logit, 1, 32));
    mx = fmaxf(mx, __shfl_xor(mx, 2, 32));
    float e  = __expf(logit - mx);
    float se = e + __shfl_xor(e, 1, 32);
    se += __shfl_xor(se, 2, 32);
    float aw = e / se;

    // lanes 0..15: (m,s) samples; weights from unclamped corners, ids from clamped
    const int ms = lane & 15, m = ms >> 2, s = ms & 3;
    float w = 0.f; int fid = 0;
    if (valid) {
        float dsrc = (float)(128 >> mid);   // MAP_HW: square maps 128>>m
        float dm   = (float)(128 >> m);
        float sx = (px + 0.5f) / dsrc * dm - 0.5f;
        float sy = (py + 0.5f) / dsrc * dm - 0.5f;
        int ix0 = (int)floorf(sx), iy0 = (int)floorf(sy);
        int cx = ix0 + (s & 1), cy = iy0 + (s >> 1);
        w = (1.f - fabsf(sx - (float)cx)) * (1.f - fabsf(sy - (float)cy));
        int dim = 128 >> m;
        int icx = min(max(cx, 0), dim - 1), icy = min(max(cy, 0), dim - 1);
        const int* imp = (m == 0) ? im0 : (m == 1) ? im1 : (m == 2) ? im2 : im3;
        fid = imp[(size_t)bid * dim * dim + icy * dim + icx];
    }
#pragma unroll
    for (int h = 0; h < NH; ++h) {
        float a = __shfl(aw, h * 4 + m, 32);       // aw[h, m]
        if (valid && lane < 16) fw[(size_t)n * 128 + ms * 8 + h] = a * w;
    }
    if (valid && lane < 16) fids[(size_t)n * 16 + ms] = fid;
}

// ---------------- Kernel 2: gather+mix -> WMMA GEMM2 -> WMMA GEMM3 ----------------
__global__ void __launch_bounds__(256)
fused_main_kernel(const float* __restrict__ act,  const float* __restrict__ pas,
                  const v16h* __restrict__ vfrag, const float* __restrict__ valb,
                  const v16h* __restrict__ ofrag, const float* __restrict__ outb,
                  const float* __restrict__ fw,   const int* __restrict__ fids,
                  float* __restrict__ out, int n_act, int n_total) {
    __shared__ _Float16 smix[64][FDIM + 8];   // 4 heads x 16 n rows (one head-group)
    __shared__ _Float16 sval[16][FDIM + 8];   // val_feats tile
    __shared__ float    sfw[16 * NS * NH];    // [nn][ms][h]
    __shared__ int      sfids[16 * NS];
    const int t = threadIdx.x;
    const int wave = t >> 5, lane = t & 31;
    const int lh = (lane >> 4) & 1, lc = lane & 15;
    const int n0 = blockIdx.x * 16;

    {   // stage per-tile mixing weights & gather ids (contiguous layouts)
        const float* gfw = fw + (size_t)n0 * 128;
#pragma unroll
        for (int i = 0; i < 8; ++i) sfw[t + i * 256] = gfw[t + i * 256];
        sfids[t] = fids[(size_t)n0 * 16 + t];
    }
    __syncthreads();

    for (int hg = 0; hg < 2; ++hg) {
        // phase 1: gather 16 rows per n and mix into 4 heads (thread = feature col)
        for (int nn = 0; nn < 16; ++nn) {
            const float* wv = &sfw[nn * 128];
            float v[NS];
#pragma unroll
            for (int ms = 0; ms < NS; ++ms) {
                int fid = sfids[nn * 16 + ms];
                const float* src = (fid < n_act) ? (act + (size_t)fid * FDIM)
                                                 : (pas + (size_t)(fid - n_act) * FDIM);
                v[ms] = src[t];
                if (hg == 0 && nn + 1 < 16) {          // prefetch next row set
                    int f2 = sfids[(nn + 1) * 16 + ms];
                    const float* s2 = (f2 < n_act) ? (act + (size_t)f2 * FDIM)
                                                   : (pas + (size_t)(f2 - n_act) * FDIM);
                    __builtin_prefetch(s2 + t, 0, 0);  // global_prefetch_b8
                }
            }
#pragma unroll
            for (int hl = 0; hl < 4; ++hl) {
                int h = hg * 4 + hl;
                float a = 0.f;
#pragma unroll
                for (int ms = 0; ms < NS; ++ms) a += wv[ms * 8 + h] * v[ms];
                smix[hl * 16 + nn][t] = (_Float16)a;
            }
        }
        __syncthreads();
        // phase 2: per-head GEMM  mixed[16,256] x valw_h^T[256,32] (wave -> (head, col-tile))
        {
            const int hl = wave >> 1, ct = wave & 1;
            const int h  = hg * 4 + hl;
            const int tile = h * 2 + ct;              // weight row-tile (16 cols)
            v8f c = {};
#pragma unroll
            for (int kk = 0; kk < 8; ++kk) {
                v16h a = frag_from_h(&smix[hl * 16 + lc][kk * 32], lh);
                v16h b = vfrag[(tile * 8 + kk) * 32 + lane];   // pre-swizzled f16
                c = __builtin_amdgcn_wmma_f32_16x16x32_f16(false, a, false, b,
                                                           (short)0, c, false, false);
            }
            const int col = tile * 16 + lc;
            const float vb = valb[col];   // softmax & bilinear weights sum to 1 -> bias passes through
#pragma unroll
            for (int r = 0; r < 8; ++r) {
                int row = (lane >> 4) * 8 + r;
                sval[row][col] = (_Float16)(c[r] + vb);
            }
        }
        __syncthreads();
    }

    // phase 3: out GEMM  val_feats[16,256] x outw^T[256,256], 16 col-tiles over 8 waves
#pragma unroll
    for (int cti = 0; cti < 2; ++cti) {
        const int ct = wave * 2 + cti;
        v8f c = {};
#pragma unroll
        for (int kk = 0; kk < 8; ++kk) {
            v16h a = frag_from_h(&sval[lc][kk * 32], lh);
            v16h b = ofrag[(ct * 8 + kk) * 32 + lane];         // pre-swizzled f16
            c = __builtin_amdgcn_wmma_f32_16x16x32_f16(false, a, false, b,
                                                       (short)0, c, false, false);
        }
        const int col = ct * 16 + lc;
        const float ob = outb[col];
#pragma unroll
        for (int r = 0; r < 8; ++r) {
            int row = (lane >> 4) * 8 + r;
            int n = n0 + row;
            if (n < n_total) out[(size_t)n * FDIM + col] = c[r] + ob;
        }
    }
}

extern "C" void kernel_launch(void* const* d_in, const int* in_sizes, int n_in,
                              void* d_out, int out_size, void* d_ws, size_t ws_size,
                              hipStream_t stream) {
    const float* act  = (const float*)d_in[0];
    const float* pas  = (const float*)d_in[1];
    const int*   bids = (const int*)d_in[2];
    const int*   mids = (const int*)d_in[3];
    const int*   xys  = (const int*)d_in[4];
    // d_in[5] map_shapes: constant square maps 128>>m, folded into the kernel
    const float* semb = (const float*)d_in[6];
    const float* attw = (const float*)d_in[7];
    const float* attb = (const float*)d_in[8];
    const float* valw = (const float*)d_in[9];
    const float* valb = (const float*)d_in[10];
    const float* outw = (const float*)d_in[11];
    const float* outb = (const float*)d_in[12];
    const int* im0 = (const int*)d_in[13];
    const int* im1 = (const int*)d_in[14];
    const int* im2 = (const int*)d_in[15];
    const int* im3 = (const int*)d_in[16];
    const int n_total = in_sizes[2];   // N = len(act_batch_ids)

    // workspace layout (all 32B aligned):
    //   fw      [N,16,8] f32            : N*128*4 bytes
    //   fids    [N,16]   i32            : N*16*4 bytes
    //   vfrag   [16][8][32][16] f16     : 131072 bytes
    //   ofrag   [16][8][32][16] f16     : 131072 bytes
    char* ws = (char*)d_ws;
    float*    fwbuf  = (float*)ws;
    int*      fidbuf = (int*)(ws + (size_t)n_total * 128 * sizeof(float));
    _Float16* vfrag  = (_Float16*)(ws + (size_t)n_total * 128 * sizeof(float)
                                      + (size_t)n_total * 16 * sizeof(int));
    _Float16* ofrag  = vfrag + 65536;
    float* out = (float*)d_out;

    dim3 blk(256);
    prep_wfrag_kernel<<<dim3(256), blk, 0, stream>>>(valw, vfrag);
    prep_wfrag_kernel<<<dim3(256), blk, 0, stream>>>(outw, ofrag);
    attn_meta_kernel<<<dim3((n_total + 7) / 8), blk, 0, stream>>>(
        act, bids, mids, xys, semb, attw, attb, im0, im1, im2, im3,
        fwbuf, fidbuf, n_total);
    fused_main_kernel<<<dim3((n_total + 15) / 16), blk, 0, stream>>>(
        act, pas, (const v16h*)vfrag, valb, (const v16h*)ofrag, outb,
        fwbuf, fidbuf, out, n_total, n_total);
}
